// MultiRepresentationFuser_v2_23141283791483
// MI455X (gfx1250) — compile-verified, matrix operand order checked by reading, not run on hardware
//
#include <hip/hip_runtime.h>
#include <math.h>

// Problem constants (match the reference)
#define HH 256
#define WW 256
#define NB 2
#define HWSZ (HH*WW)            // 65536
#define NPIX (NB*HWSZ)          // 131072 pixels
#define NCATCOLS 108            // 72 off + 36 mask (padded to 128 for alignment)

// ---------------- CDNA5 WMMA types ----------------
typedef __attribute__((ext_vector_type(16))) __bf16 v16bf;
typedef __attribute__((ext_vector_type(8)))  float  v8f;

union FragBF { v16bf v; unsigned u[8]; };

// ---------------- helpers ----------------
__device__ __forceinline__ unsigned short f2bf(float f) {
  unsigned u = __float_as_uint(f);
  unsigned r = u + 0x7FFFu + ((u >> 16) & 1u);   // RNE
  return (unsigned short)(r >> 16);
}
__device__ __forceinline__ float geluf(float x) {        // tanh approx (jax default)
  float x3 = x * x * x;
  return 0.5f * x * (1.0f + tanhf(0.7978845608028654f * (x + 0.044715f * x3)));
}
__device__ __forceinline__ float siluf(float x) {
  return x / (1.0f + __expf(-x));
}
// K index pattern for 16-bit WMMA A/B fragments (16x32), per CDNA5 ISA 7.12.2
__device__ __forceinline__ int k_pattern(int p, int half) {
  return (p < 4) ? (half * 8 + 2 * p) : (16 + half * 8 + 2 * (p - 4));
}

// ---------------- Poisson (div + Jacobi) ----------------
__global__ void k_fill0(float* p, int n) {
  int i = blockIdx.x * blockDim.x + threadIdx.x;
  if (i < n) p[i] = 0.f;
}

__global__ void k_div(const float* __restrict__ x, float* __restrict__ divg) {
  int p = blockIdx.x * blockDim.x + threadIdx.x;
  if (p >= NPIX) return;
  int b = p >> 16, rem = p & 65535, y = rem >> 8, xx = rem & 255;
  const float* d0p = x + (size_t)(b * 9 + 5) * HWSZ;   // sd1 ch0
  const float* d1p = x + (size_t)(b * 9 + 6) * HWSZ;   // sd1 ch1
  int idx = y * WW + xx;
  // gx = -Ix1 = -0.5*(d0+d1) ; gy = -Iy1 = 0.5*(d0-d1)
  float gx  = -0.5f * (d0p[idx] + d1p[idx]);
  float gy  =  0.5f * (d0p[idx] - d1p[idx]);
  // unconditional clamped loads; mask values (avoid predicated loads)
  int jl = idx - (xx > 0 ? 1 : 0);
  int ju = idx - (y  > 0 ? WW : 0);
  float gxl = -0.5f * (d0p[jl] + d1p[jl]);
  float gyu =  0.5f * (d0p[ju] - d1p[ju]);
  gxl = (xx > 0) ? gxl : 0.f;
  gyu = (y  > 0) ? gyu : 0.f;
  divg[p] = (gx - gxl) + (gy - gyu);
}

__global__ void k_jacobi(const float* __restrict__ u, const float* __restrict__ divg,
                         float* __restrict__ un) {
  int p = blockIdx.x * blockDim.x + threadIdx.x;
  if (p >= NPIX) return;
  int base = p & ~65535;
  int rem = p & 65535, y = rem >> 8, xx = rem & 255;
  const float* ub = u + base;
  float up_ = ub[(y > 0      ? rem - WW : rem)];
  float dn  = ub[(y < HH - 1 ? rem + WW : rem)];
  float lf  = ub[(xx > 0     ? rem - 1  : rem)];
  float rt  = ub[(xx < WW - 1? rem + 1  : rem)];
  un[p] = (up_ + dn + lf + rt - divg[p]) * 0.25f;
}

// ---------------- conv1: 7->32 3x3 + silu, NCHW in -> bf16 NHWC out ---------
__global__ void k_conv1(const float* __restrict__ x, const float* __restrict__ w,
                        const float* __restrict__ bias, unsigned short* __restrict__ h1bf) {
  __shared__ float sw[32 * 7 * 9];
  for (int i = threadIdx.x; i < 32 * 7 * 9; i += blockDim.x) sw[i] = w[i];
  __syncthreads();
  int p = blockIdx.x * blockDim.x + threadIdx.x;
  if (p >= NPIX) return;
  int b = p >> 16, rem = p & 65535, y = rem >> 8, xx = rem & 255;
  float acc[32];
#pragma unroll
  for (int oc = 0; oc < 32; ++oc) acc[oc] = bias[oc];
  const float* inb = x + (size_t)b * 9 * HWSZ;   // first 7 channels = [img0, td, sd1]
  for (int ic = 0; ic < 7; ++ic) {
    float v[9];
    const float* ip = inb + (size_t)ic * HWSZ;
#pragma unroll
    for (int t = 0; t < 9; ++t) {
      int yy = y + t / 3 - 1, xc = xx + t % 3 - 1;
      bool ok = (yy >= 0 && yy < HH && xc >= 0 && xc < WW);
      int yc = min(max(yy, 0), HH - 1), xcc = min(max(xc, 0), WW - 1);
      float lv = ip[yc * WW + xcc];              // unconditional safe load
      v[t] = ok ? lv : 0.f;                      // value select, not load predication
    }
    for (int oc = 0; oc < 32; ++oc) {
      const float* wp = sw + (oc * 7 + ic) * 9;
      float s = 0.f;
#pragma unroll
      for (int t = 0; t < 9; ++t) s += v[t] * wp[t];
      acc[oc] += s;
    }
  }
  unsigned short* ob = h1bf + (size_t)p * 32;
#pragma unroll
  for (int oc = 0; oc < 32; ++oc) ob[oc] = f2bf(siluf(acc[oc]));
}

// ---------------- conv2 weight pack: K=(tap,ic), per-lane WMMA B layout -----
// pack index: (((tap*2 + ntile)*32 + lane)*8 + p) -> dword {bf16(ic), bf16(ic+1)}
__global__ void k_pack_conv2(const float* __restrict__ rw2, unsigned* __restrict__ pack) {
  int total = 9 * 2 * 32 * 8;
  for (int i = blockIdx.x * blockDim.x + threadIdx.x; i < total; i += gridDim.x * blockDim.x) {
    int pp = i & 7, lane = (i >> 3) & 31, nt = (i >> 8) & 1, t = i >> 9;
    int oc = nt * 16 + (lane & 15), half = lane >> 4;
    int ic = k_pattern(pp, half);                         // 0..30 even
    float w0 = rw2[(size_t)(oc * 32 + ic) * 9 + t];       // OIHW: (oc,ic,ky,kx)
    float w1 = rw2[(size_t)(oc * 32 + ic + 1) * 9 + t];
    pack[i] = (unsigned)f2bf(w0) | ((unsigned)f2bf(w1) << 16);
  }
}

// ---------------- conv2: 32->32 3x3 + silu as WMMA implicit GEMM ------------
// A: 16 pixels x 32 ic (one tap per K-step, 9 K-steps); B: packed weights.
// h1 is bf16 NHWC (32 ch = 16 dwords per pixel); h2 out is fp32 NHWC.
// Halo handling: unconditional clamped-address loads, zero via bitwise mask so
// loads stay coalesced (no exec-mask predication around memory ops).
__global__ void k_conv2_wmma(const unsigned* __restrict__ h1bf32,
                             const unsigned* __restrict__ Bpack,
                             const float* __restrict__ bias,
                             float* __restrict__ h2) {
  int lane = threadIdx.x & 31, wave = threadIdx.x >> 5;
  int rowTile = blockIdx.x * 4 + wave;
  if (rowTile * 16 >= NPIX) return;                      // wave-uniform
  int half = lane >> 4;
  int r = rowTile * 16 + (lane & 15);
  int b = r >> 16, rem = r & 65535, y = rem >> 8, xx = rem & 255;
  v8f acc0 = {}, acc1 = {};
#pragma unroll
  for (int t = 0; t < 9; ++t) {
    int yy = y + t / 3 - 1, xc = xx + t % 3 - 1;
    bool ok = (yy >= 0 && yy < HH && xc >= 0 && xc < WW);
    int yc = min(max(yy, 0), HH - 1), xcc = min(max(xc, 0), WW - 1);  // safe addr
    unsigned msk = ok ? 0xFFFFFFFFu : 0u;
    const unsigned* ap = h1bf32 + ((size_t)(b * HWSZ + yc * WW + xcc)) * 16;
    FragBF a;
#pragma unroll
    for (int p = 0; p < 8; ++p) {
      int k0 = k_pattern(p, half);                       // ic pair within tap
      a.u[p] = ap[k0 >> 1] & msk;                        // load always, mask value
    }
    FragBF b0, b1;
    const unsigned* bp0 = Bpack + ((size_t)(t * 2 + 0) * 32 + lane) * 8;
    const unsigned* bp1 = Bpack + ((size_t)(t * 2 + 1) * 32 + lane) * 8;
#pragma unroll
    for (int p = 0; p < 8; ++p) { b0.u[p] = bp0[p]; b1.u[p] = bp1[p]; }
    acc0 = __builtin_amdgcn_wmma_f32_16x16x32_bf16(false, a.v, false, b0.v, (short)0, acc0, false, false);
    acc1 = __builtin_amdgcn_wmma_f32_16x16x32_bf16(false, a.v, false, b1.v, (short)0, acc1, false, false);
  }
  // epilogue: bias + silu -> fp32 NHWC
  {
    int col = lane & 15;
    float bia = bias[col];
#pragma unroll
    for (int v = 0; v < 8; ++v) {
      int rr = rowTile * 16 + v + 8 * half;
      h2[(size_t)rr * 32 + col] = siluf(acc0[v] + bia);
    }
  }
  {
    int col = 16 + (lane & 15);
    float bia = bias[col];
#pragma unroll
    for (int v = 0; v < 8; ++v) {
      int rr = rowTile * 16 + v + 8 * half;
      h2[(size_t)rr * 32 + col] = siluf(acc1[v] + bia);
    }
  }
}

// ---------------- conv3: 32->3 3x3, fp32 NHWC in -> recon pixel-major x3 ----
__global__ void k_conv3(const float* __restrict__ h2, const float* __restrict__ w,
                        const float* __restrict__ bias, float* __restrict__ recon) {
  __shared__ float sw[3 * 32 * 9];
  for (int i = threadIdx.x; i < 3 * 32 * 9; i += blockDim.x) sw[i] = w[i];
  __syncthreads();
  int p = blockIdx.x * blockDim.x + threadIdx.x;
  if (p >= NPIX) return;
  int b = p >> 16, rem = p & 65535, y = rem >> 8, xx = rem & 255;
  float a0 = bias[0], a1 = bias[1], a2 = bias[2];
  for (int t = 0; t < 9; ++t) {
    int yy = y + t / 3 - 1, xc = xx + t % 3 - 1;
    if (yy < 0 || yy >= HH || xc < 0 || xc >= WW) continue;
    const float* np = h2 + ((size_t)(b * HWSZ + yy * WW + xc)) * 32;
    for (int ic = 0; ic < 32; ++ic) {
      float v = np[ic];
      a0 += v * sw[(0 * 32 + ic) * 9 + t];
      a1 += v * sw[(1 * 32 + ic) * 9 + t];
      a2 += v * sw[(2 * 32 + ic) * 9 + t];
    }
  }
  float* rp = recon + (size_t)p * 3;
  rp[0] = a0; rp[1] = a1; rp[2] = a2;
}

// ---------------- pointwise linears (K=7 / K=4, VALU) -----------------------
__global__ void k_pointwise_in(const float* __restrict__ x, const float* __restrict__ gray,
                               const float* __restrict__ recon,
                               const float* __restrict__ Win,  const float* __restrict__ bin,
                               const float* __restrict__ Waop, const float* __restrict__ baop,
                               float* __restrict__ xfeat, float* __restrict__ apre) {
  int p = blockIdx.x * blockDim.x + threadIdx.x;
  if (p >= NPIX) return;
  int b = p >> 16, rem = p & 65535;
  const float* xb = x + (size_t)b * 9 * HWSZ + rem;
  float core[7], guide[4];
  core[0] = xb[0];          core[1] = xb[HWSZ];       core[2] = xb[2 * HWSZ];   // img0
  core[3] = gray[p];                                                             // poisson gray
  const float* rp = recon + (size_t)p * 3;
  core[4] = rp[0];          core[5] = rp[1];          core[6] = rp[2];          // recon
  guide[0] = xb[7 * HWSZ];  guide[1] = xb[8 * HWSZ];                            // sd0
  guide[2] = xb[5 * HWSZ];  guide[3] = xb[6 * HWSZ];                            // sd1
  float* xo = xfeat + (size_t)p * 64;
  float* ao = apre  + (size_t)p * 64;
  for (int j = 0; j < 64; ++j) {
    float s = bin[j];
#pragma unroll
    for (int i = 0; i < 7; ++i) s += core[i] * Win[i * 64 + j];
    xo[j] = s;
    float t = baop[j];
#pragma unroll
    for (int i = 0; i < 4; ++i) t += guide[i] * Waop[i * 64 + j];
    ao[j] = geluf(t);
  }
}

// ---------------- depthwise 3x3 (NHWC, zero pad) + gelu -> bf16 -------------
__global__ void k_dwconv(const float* __restrict__ apre, const float* __restrict__ dw,
                         unsigned short* __restrict__ afeat_bf) {
  __shared__ float sdw[9 * 64];
  for (int i = threadIdx.x; i < 576; i += blockDim.x) sdw[i] = dw[i];
  __syncthreads();
  int p = blockIdx.x * blockDim.x + threadIdx.x;
  if (p >= NPIX) return;
  int b = p >> 16, rem = p & 65535, y = rem >> 8, xx = rem & 255;
  float acc[64];
#pragma unroll
  for (int c = 0; c < 64; ++c) acc[c] = 0.f;
  for (int t = 0; t < 9; ++t) {
    int yy = y + t / 3 - 1, xc = xx + t % 3 - 1;
    if (yy < 0 || yy >= HH || xc < 0 || xc >= WW) continue;   // coarse skip (64ch of work)
    const float* np = apre + ((size_t)(b * HWSZ + yy * WW + xc)) * 64;
    const float* wp = sdw + t * 64;
    for (int c = 0; c < 64; ++c) acc[c] += np[c] * wp[c];
  }
  unsigned short* op = afeat_bf + (size_t)p * 64;
  for (int c = 0; c < 64; ++c) op[c] = f2bf(geluf(acc[c]));
}

// ---------------- weight pre-pack into per-lane WMMA B layout ---------------
// pack index: (((ntile*2 + kstep)*32 + lane)*8 + p) -> dword {bf16(k), bf16(k+1)}
__global__ void k_pack_cat(const float* __restrict__ Woff, const float* __restrict__ Wmask,
                           unsigned* __restrict__ pack) {
  int total = 8 * 2 * 32 * 8;
  for (int i = blockIdx.x * blockDim.x + threadIdx.x; i < total; i += gridDim.x * blockDim.x) {
    int pp = i & 7, lane = (i >> 3) & 31, ks = (i >> 8) & 1, nt = i >> 9;
    int col = nt * 16 + (lane & 15), half = lane >> 4;
    int k = ks * 32 + k_pattern(pp, half);
    float w0, w1;
    if (col < 72)        { w0 = Woff[k * 72 + col];        w1 = Woff[(k + 1) * 72 + col]; }
    else if (col < 108)  { w0 = Wmask[k * 36 + col - 72];  w1 = Wmask[(k + 1) * 36 + col - 72]; }
    else                 { w0 = 0.f;                       w1 = 0.f; }
    pack[i] = (unsigned)f2bf(w0) | ((unsigned)f2bf(w1) << 16);
  }
}

__global__ void k_pack_out(const float* __restrict__ Wout, unsigned* __restrict__ pack) {
  int total = 4 * 2 * 32 * 8;
  for (int i = blockIdx.x * blockDim.x + threadIdx.x; i < total; i += gridDim.x * blockDim.x) {
    int pp = i & 7, lane = (i >> 3) & 31, ks = (i >> 8) & 1, nt = i >> 9;
    int col = nt * 16 + (lane & 15), half = lane >> 4;
    int k = ks * 32 + k_pattern(pp, half);
    pack[i] = (unsigned)f2bf(Wout[k * 64 + col]) | ((unsigned)f2bf(Wout[(k + 1) * 64 + col]) << 16);
  }
}

// ---------------- WMMA GEMM helpers ----------------
__device__ __forceinline__ void load_a_frags(FragBF& a0, FragBF& a1,
                                             const unsigned* __restrict__ A32,
                                             int row, int half) {
  const unsigned* rp = A32 + (size_t)row * 32;   // 64 bf16 = 32 dwords per row
#pragma unroll
  for (int p = 0; p < 8; ++p) {
    int k0 = k_pattern(p, half);                 // even -> contiguous dword pair
    a0.u[p] = rp[k0 >> 1];
    a1.u[p] = rp[(k0 >> 1) + 16];                // K base 32
  }
}

// GEMM: offmask[N x 128] = afeat[N x 64] @ Wcat[64 x 108pad128] (+bias)
__global__ void k_gemm_offmask(const unsigned* __restrict__ A32,
                               const unsigned* __restrict__ Bpack,
                               const float* __restrict__ boff, const float* __restrict__ bmask,
                               float* __restrict__ offmask) {
  int lane = threadIdx.x & 31, wave = threadIdx.x >> 5;
  int rowTile = blockIdx.x * 4 + wave;
  if (rowTile * 16 >= NPIX) return;              // wave-uniform
  int half = lane >> 4;
  int row = rowTile * 16 + (lane & 15);
  FragBF a0, a1;
  load_a_frags(a0, a1, A32, row, half);
  for (int nt = 0; nt < 8; ++nt) {
    FragBF b0, b1;
    const unsigned* bp0 = Bpack + ((size_t)(nt * 2 + 0) * 32 + lane) * 8;
    const unsigned* bp1 = Bpack + ((size_t)(nt * 2 + 1) * 32 + lane) * 8;
#pragma unroll
    for (int p = 0; p < 8; ++p) { b0.u[p] = bp0[p]; b1.u[p] = bp1[p]; }
    v8f acc = {};
    acc = __builtin_amdgcn_wmma_f32_16x16x32_bf16(false, a0.v, false, b0.v, (short)0, acc, false, false);
    acc = __builtin_amdgcn_wmma_f32_16x16x32_bf16(false, a1.v, false, b1.v, (short)0, acc, false, false);
    int col = nt * 16 + (lane & 15);
    if (col < NCATCOLS) {
      float bia = (col < 72) ? boff[col] : bmask[col - 72];
#pragma unroll
      for (int v = 0; v < 8; ++v) {
        int r = rowTile * 16 + v + 8 * half;
        offmask[(size_t)r * 128 + col] = acc[v] + bia;
      }
    }
  }
}

// GEMM: d_out (NCHW) = accb[N x 64] @ Wout[64 x 64] + bout, transposed store
__global__ void k_gemm_out(const unsigned* __restrict__ A32,
                           const unsigned* __restrict__ Bpack,
                           const float* __restrict__ bout,
                           float* __restrict__ out) {
  int lane = threadIdx.x & 31, wave = threadIdx.x >> 5;
  int rowTile = blockIdx.x * 4 + wave;
  if (rowTile * 16 >= NPIX) return;
  int half = lane >> 4;
  int row = rowTile * 16 + (lane & 15);
  FragBF a0, a1;
  load_a_frags(a0, a1, A32, row, half);
  for (int nt = 0; nt < 4; ++nt) {
    FragBF b0, b1;
    const unsigned* bp0 = Bpack + ((size_t)(nt * 2 + 0) * 32 + lane) * 8;
    const unsigned* bp1 = Bpack + ((size_t)(nt * 2 + 1) * 32 + lane) * 8;
#pragma unroll
    for (int p = 0; p < 8; ++p) { b0.u[p] = bp0[p]; b1.u[p] = bp1[p]; }
    v8f acc = {};
    acc = __builtin_amdgcn_wmma_f32_16x16x32_bf16(false, a0.v, false, b0.v, (short)0, acc, false, false);
    acc = __builtin_amdgcn_wmma_f32_16x16x32_bf16(false, a1.v, false, b1.v, (short)0, acc, false, false);
    int c = nt * 16 + (lane & 15);
    float bia = bout[c];
#pragma unroll
    for (int v = 0; v < 8; ++v) {
      int r = rowTile * 16 + v + 8 * half;        // pixel index
      int b = r >> 16, hw = r & 65535;
      out[((size_t)(b * 64 + c)) * HWSZ + hw] = acc[v] + bia;   // (B,64,H,W)
    }
  }
}

// ---------------- deformable sampling (softmax + bilinear gather) -----------
__global__ void k_sample(const float* __restrict__ xfeat, const float* __restrict__ offmask,
                         unsigned short* __restrict__ accbf) {
  int t = blockIdx.x * blockDim.x + threadIdx.x;
  if (t >= NPIX * 4) return;
  int g = t & 3;
  int p = t >> 2;
  int b = p >> 16, rem = p & 65535, y = rem >> 8, xx = rem & 255;
  const float* om = offmask + (size_t)p * 128;
  // softmax over 9 mask logits of group g
  float lg[9], mx = -1e30f;
#pragma unroll
  for (int k = 0; k < 9; ++k) { lg[k] = om[72 + g * 9 + k]; mx = fmaxf(mx, lg[k]); }
  float sum = 0.f;
#pragma unroll
  for (int k = 0; k < 9; ++k) { lg[k] = __expf(lg[k] - mx); sum += lg[k]; }
  float inv = 1.0f / sum;
  float acc[16];
#pragma unroll
  for (int c = 0; c < 16; ++c) acc[c] = 0.f;
  const float* xb = xfeat + ((size_t)b * HWSZ) * 64 + g * 16;
#pragma unroll
  for (int k = 0; k < 9; ++k) {
    float m  = lg[k] * inv;
    float oy = om[g * 18 + k * 2 + 0];
    float ox = om[g * 18 + k * 2 + 1];
    float ly = (float)y  + (float)(k / 3 - 1) + oy;
    float lx = (float)xx + (float)(k % 3 - 1) + ox;
    float y0f = floorf(ly), x0f = floorf(lx);
    float wy = ly - y0f, wx = lx - x0f;
    int y0 = (int)y0f, x0 = (int)x0f;
#pragma unroll
    for (int dy = 0; dy < 2; ++dy) {
      int yi = y0 + dy;
      float wgy = dy ? wy : 1.f - wy;
      bool vy = (yi >= 0 && yi < HH);
      int yc = min(max(yi, 0), HH - 1);
#pragma unroll
      for (int dx = 0; dx < 2; ++dx) {
        int xi = x0 + dx;
        float wgx = dx ? wx : 1.f - wx;
        bool vx = (xi >= 0 && xi < WW);
        int xc = min(max(xi, 0), WW - 1);
        float wv = m * wgy * wgx * ((vy && vx) ? 1.f : 0.f);
        const float* vp = xb + (size_t)(yc * WW + xc) * 64;
#pragma unroll
        for (int c = 0; c < 16; ++c) acc[c] += wv * vp[c];
      }
    }
  }
  unsigned short* op = accbf + (size_t)p * 64 + g * 16;
#pragma unroll
  for (int c = 0; c < 16; ++c) op[c] = f2bf(acc[c]);
}

// ---------------- launcher ----------------
extern "C" void kernel_launch(void* const* d_in, const int* in_sizes, int n_in,
                              void* d_out, int out_size, void* d_ws, size_t ws_size,
                              hipStream_t stream) {
  (void)in_sizes; (void)n_in; (void)out_size; (void)ws_size;
  const float* x     = (const float*)d_in[0];
  const float* rw1   = (const float*)d_in[1];
  const float* rb1   = (const float*)d_in[2];
  const float* rw2   = (const float*)d_in[3];
  const float* rb2   = (const float*)d_in[4];
  const float* rw3   = (const float*)d_in[5];
  const float* rb3   = (const float*)d_in[6];
  const float* Win   = (const float*)d_in[7];
  const float* bin   = (const float*)d_in[8];
  const float* Waop  = (const float*)d_in[9];
  const float* baop  = (const float*)d_in[10];
  const float* dw    = (const float*)d_in[11];
  const float* Woff  = (const float*)d_in[12];
  const float* boff  = (const float*)d_in[13];
  const float* Wmask = (const float*)d_in[14];
  const float* bmask = (const float*)d_in[15];
  const float* Wout  = (const float*)d_in[16];
  const float* bout  = (const float*)d_in[17];
  float* out = (float*)d_out;

  char* ws = (char*)d_ws;
  size_t off = 0;
  auto alloc = [&](size_t bytes) -> void* {
    void* pp = ws + off;
    off += (bytes + 255) & ~(size_t)255;
    return pp;
  };
  float*          divg    = (float*)alloc((size_t)NPIX * 4);
  float*          u_a     = (float*)alloc((size_t)NPIX * 4);
  float*          u_b     = (float*)alloc((size_t)NPIX * 4);
  unsigned short* h1bf    = (unsigned short*)alloc((size_t)NPIX * 32 * 2);   // bf16 NHWC
  float*          h2      = (float*)alloc((size_t)NPIX * 32 * 4);            // fp32 NHWC
  float*          recon   = (float*)alloc((size_t)NPIX * 3 * 4);             // pixel-major x3
  float*          xfeat   = (float*)alloc((size_t)NPIX * 64 * 4);
  float*          apre    = (float*)alloc((size_t)NPIX * 64 * 4);
  unsigned short* afeat   = (unsigned short*)alloc((size_t)NPIX * 64 * 2);
  float*          offm    = (float*)alloc((size_t)NPIX * 128 * 4);
  unsigned short* accb    = (unsigned short*)alloc((size_t)NPIX * 64 * 2);
  unsigned*       packC2  = (unsigned*)alloc(4608 * 4);
  unsigned*       packCat = (unsigned*)alloc(4096 * 4);
  unsigned*       packOut = (unsigned*)alloc(2048 * 4);

  dim3 blk(256);
  dim3 gpix(NPIX / 256);

  // Poisson gray
  k_fill0<<<gpix, blk, 0, stream>>>(u_a, NPIX);
  k_div<<<gpix, blk, 0, stream>>>(x, divg);
  float* prev = u_a; float* next = u_b;
  for (int it = 0; it < 20; ++it) {
    k_jacobi<<<gpix, blk, 0, stream>>>(prev, divg, next);
    float* tmp = prev; prev = next; next = tmp;
  }
  const float* gray = prev;

  // recon conv stack: VALU 7->32 (bf16 NHWC), WMMA 32->32, VALU 32->3
  k_conv1<<<gpix, blk, 0, stream>>>(x, rw1, rb1, h1bf);
  k_pack_conv2<<<5, 256, 0, stream>>>(rw2, packC2);
  k_conv2_wmma<<<NPIX / 64, 128, 0, stream>>>((const unsigned*)h1bf, packC2, rb2, h2);
  k_conv3<<<gpix, blk, 0, stream>>>(h2, rw3, rb3, recon);

  // dcnv3 front-end
  k_pointwise_in<<<gpix, blk, 0, stream>>>(x, gray, recon, Win, bin, Waop, baop, xfeat, apre);
  k_dwconv<<<gpix, blk, 0, stream>>>(apre, dw, afeat);

  // pack weights into WMMA lane layout
  k_pack_cat<<<4, 256, 0, stream>>>(Woff, Wmask, packCat);
  k_pack_out<<<2, 256, 0, stream>>>(Wout, packOut);

  // WMMA GEMMs + sampling
  k_gemm_offmask<<<NPIX / 64, 128, 0, stream>>>((const unsigned*)afeat, packCat, boff, bmask, offm);
  k_sample<<<(NPIX * 4) / 256, blk, 0, stream>>>(xfeat, offm, accb);
  k_gemm_out<<<NPIX / 64, 128, 0, stream>>>((const unsigned*)accb, packOut, bout, out);
}
